// PointsProposalGenerator_24343874633951
// MI455X (gfx1250) — matrix operand order repaired and checked by reading, not compiled, not run on hardware
//
#include <hip/hip_runtime.h>
#include <hip/hip_bf16.h>
#include <stdint.h>

typedef __attribute__((ext_vector_type(16))) __bf16          v16bf;
typedef __attribute__((ext_vector_type(8)))  float           v8f;
typedef __attribute__((ext_vector_type(16))) unsigned short  v16us;
typedef __attribute__((ext_vector_type(8)))  unsigned short  v8us;
typedef int v4i_ __attribute__((__vector_size__(16)));       // matches builtin param type

#define Nimg   2
#define Cch    128
#define Hf     256
#define Wf     256
#define Hp     258            // padded (1-pixel zero halo)
#define Wp     258
#define HW     (Hf*Wf)
#define Ktot   (Cch*9)        // 1152
#define TP     64             // pixels per workgroup tile
#define NT     4              // 16-wide N tiles per wave
#define KSTEPS 36             // 9 taps * 4 c_in chunks
#define PRE_K  1000
#define POST_K 256
#define CAND   1024
#define OUT_ELEMS (Nimg*HW*5)                  // 655360
#define KB_OFF    OUT_ELEMS
#define KS_OFF    (OUT_ELEMS + Nimg*POST_K*4)

#if __has_builtin(__builtin_amdgcn_global_load_async_to_lds_b128) && \
    __has_builtin(__builtin_amdgcn_s_wait_asynccnt)
#define USE_ASYNC 1
#else
#define USE_ASYNC 0
#endif

// ---------- helpers ----------
__device__ __forceinline__ unsigned short f2bf(float f) {
  unsigned int u = __builtin_bit_cast(unsigned int, f);
  u += 0x7fffu + ((u >> 16) & 1u);          // round-to-nearest-even
  return (unsigned short)(u >> 16);
}

// 16B global -> LDS copy; async-to-LDS on gfx1250 toolchains that expose it
__device__ __forceinline__ void copy_b128(const unsigned short* g, unsigned short* l) {
#if USE_ASYNC
  __builtin_amdgcn_global_load_async_to_lds_b128(
      (__attribute__((address_space(1))) v4i_*)g,
      (__attribute__((address_space(3))) v4i_*)l, 0, 0);
#else
  *(uint4*)l = *(const uint4*)g;
#endif
}
__device__ __forceinline__ void copy_wait() {
#if USE_ASYNC
  __builtin_amdgcn_s_wait_asynccnt(0);
#endif
}

// A-matrix 16-bit frag (16x32): lane group g holds K = {8g..8g+7} U {16+8g..16+8g+7}
__device__ __forceinline__ v16bf load_fragA(const unsigned short* rowbase, int g) {
  v8us lo = *(const v8us*)(rowbase + 8*g);
  v8us hi = *(const v8us*)(rowbase + 16 + 8*g);
  v16us t;
#pragma unroll
  for (int i = 0; i < 8; ++i) { t[i] = lo[i]; t[8+i] = hi[i]; }
  return __builtin_bit_cast(v16bf, t);
}

// B-matrix 16-bit frag (32x16): lane group g holds K = {8g..8g+7} U {16+8g..16+8g+7}
__device__ __forceinline__ v16bf load_fragB(const unsigned short* colbase, int g) {
  v8us lo = *(const v8us*)(colbase + 8*g);
  v8us hi = *(const v8us*)(colbase + 16 + 8*g);
  v16us t;
#pragma unroll
  for (int i = 0; i < 8; ++i) { t[i] = lo[i]; t[8+i] = hi[i]; }
  return __builtin_bit_cast(v16bf, t);
}

__device__ __forceinline__ unsigned int fkey(float f) {
  unsigned int u = __builtin_bit_cast(unsigned int, f);
  return u ^ ((u >> 31) ? 0xFFFFFFFFu : 0x80000000u);
}

// ---------- prep kernels ----------
// NCHW f32 -> padded NHWC bf16 (zero halo so im2col staging is branch-free)
__global__ void k_prep_x(const float* __restrict__ x, unsigned short* __restrict__ xbp) {
  int t = blockIdx.x * 256 + threadIdx.x;
  if (t >= Nimg*Hp*Wp*Cch) return;
  int c = t & (Cch - 1);
  int r = t >> 7;
  int ww = r % Wp; r /= Wp;
  int hh = r % Hp; int n = r / Hp;
  float v = 0.f;
  if (hh >= 1 && hh <= Hf && ww >= 1 && ww <= Wf)
    v = x[(((size_t)n*Cch + c)*Hf + (hh - 1))*Wf + (ww - 1)];
  xbp[t] = f2bf(v);
}

__global__ void k_prep_w(const float* __restrict__ Wc, unsigned short* __restrict__ Wt) {
  int t = blockIdx.x * 256 + threadIdx.x;
  if (t >= Cch*Ktot) return;
  int k  = t % Ktot;
  int co = t / Ktot;
  int tap = k >> 7;                 // K = tap*128 + c_in
  int ci  = k & 127;
  int kh = tap / 3, kw = tap % 3;
  Wt[t] = f2bf(Wc[(((size_t)co*Cch + ci)*3 + kh)*3 + kw]);
}

__global__ void k_prep_w19(const float* __restrict__ Wl, const float* __restrict__ Wo,
                           unsigned short* __restrict__ W19) {
  int t = blockIdx.x * 256 + threadIdx.x;
  if (t >= 32*Cch) return;
  int c = t & 127, o = t >> 7;
  float v = 0.f;
  if (o == 0) v = Wl[c];
  else if (o <= 18) v = Wo[(size_t)(o - 1)*Cch + c];
  W19[t] = f2bf(v);
}

// ---------- fused conv3x3 + relu + 1x1 proj + decode ----------
__global__ __launch_bounds__(256) void k_conv_fused(
    const unsigned short* __restrict__ xbp,  // [N][Hp][Wp][C] bf16, zero halo
    const unsigned short* __restrict__ Wt,   // [c_out][Ktot] bf16
    const unsigned short* __restrict__ W19,  // [32][128] bf16
    const float* __restrict__ b_conv,
    const float* __restrict__ b_off,
    const float* __restrict__ b_log,
    const int* __restrict__ image_h_p,
    const int* __restrict__ image_w_p,
    float* __restrict__ out)                 // [N][HW][5]
{
  __shared__ __align__(16) unsigned short ldsA[2][Cch*32];  // weight chunk   2x8KB
  __shared__ __align__(16) unsigned short ldsB[2][TP*32];   // x chunk        2x4KB
  __shared__ __align__(16) unsigned short ldsW19[32*Cch];   // proj weights     8KB
  __shared__ __align__(16) unsigned short ldsT[TP*Cch];     // t tile [pix][c] 16KB
  __shared__ float ldsP[32*TP];                             // proj [o][pix]    8KB

  const int tid  = threadIdx.x;
  const int wave = tid >> 5, lane = tid & 31;
  const int row  = lane & 15, g = lane >> 4;
  const int wblk = blockIdx.x, h = blockIdx.y, n = blockIdx.z;

  for (int i = tid; i < (32*Cch)/8; i += 256)
    ((uint4*)ldsW19)[i] = ((const uint4*)W19)[i];

  // per-thread staging roles (fixed for all K-steps)
  const int am = tid >> 1, ah = tid & 1;                 // A: row, 16-ushort half
  const int bp = tid >> 2, bq = tid & 3;                 // B: pixel, 8-ushort quarter
  const unsigned short* Abase = Wt + (size_t)am*Ktot + ah*16;
  const unsigned short* Bbase = xbp + (((size_t)n*Hp + h)*Wp + (size_t)(wblk*TP + bp))*Cch + bq*8;

  v8f acc[NT] = {};

  // stage(it): it = tap*4 + ccq ; K chunk = tap*128 + ccq*32
  auto stage = [&](int it, int buf) {
    const int tap = it >> 2, cc = (it & 3) * 32;
    const unsigned short* as = Abase + tap*Cch + cc;
    copy_b128(as,     &ldsA[buf][am*32 + ah*16]);
    copy_b128(as + 8, &ldsA[buf][am*32 + ah*16 + 8]);
    const int kh = tap / 3, kw = tap % 3;                // padded offsets 0..2
    copy_b128(Bbase + ((size_t)kh*Wp + kw)*Cch + cc, &ldsB[buf][bp*32 + bq*8]);
  };

  stage(0, 0);
  for (int it = 0; it < KSTEPS; ++it) {
    const int buf = it & 1;
    copy_wait();
    __syncthreads();
    if (it + 1 < KSTEPS) stage(it + 1, buf ^ 1);
    v16bf a = load_fragA(&ldsA[buf][(wave*16 + row)*32], g);
#pragma unroll
    for (int nt = 0; nt < NT; ++nt) {
      v16bf b = load_fragB(&ldsB[buf][(nt*16 + row)*32], g);
      acc[nt] = __builtin_amdgcn_wmma_f32_16x16x32_bf16(false, a, false, b,
                                                        (short)0, acc[nt], false, false);
    }
  }

  // bias + relu; stash t tile as bf16 [pix][c] (B-layout for projection GEMM)
#pragma unroll
  for (int r = 0; r < 8; ++r) {
    const int crow = wave*16 + g*8 + r;      // C/D layout: VGPR r -> M = r + 8*g
    const float bb = b_conv[crow];
#pragma unroll
    for (int nt = 0; nt < NT; ++nt) {
      float v = acc[nt][r] + bb; v = v > 0.f ? v : 0.f;
      ldsT[(nt*16 + row)*Cch + crow] = f2bf(v);
    }
  }
  __syncthreads();

  // projection: proj[32 x 64pix] = W19[32x128] * t[128 x 64pix]; 8 tiles, 8 waves
  {
    const int mt = wave >> 2, nt = wave & 3;
    v8f p = {};
#pragma unroll
    for (int kc = 0; kc < Cch; kc += 32) {
      v16bf a2 = load_fragA(&ldsW19[(mt*16 + row)*Cch + kc], g);
      v16bf b2 = load_fragB(&ldsT[(nt*16 + row)*Cch + kc], g);
      p = __builtin_amdgcn_wmma_f32_16x16x32_bf16(false, a2, false, b2,
                                                  (short)0, p, false, false);
    }
#pragma unroll
    for (int r = 0; r < 8; ++r)
      ldsP[(mt*16 + g*8 + r)*TP + nt*16 + row] = p[r];
  }
  __syncthreads();

  // decode: exp offsets, add grid, clip, min/max over 9 points
  if (tid < TP) {
    const int pw = wblk*TP + tid;
    const float img_w = (float)(*image_w_p), img_h = (float)(*image_h_p);
    const float scale = img_w / (float)Wf;
    const float xs = (img_h - 1.f) * (float)pw / (float)(Wf - 1);
    const float ys = (img_w - 1.f) * (float)h  / (float)(Hf - 1);
    const float score = ldsP[0*TP + tid] + b_log[0];
    float mnx = 3.402823466e38f, mny = 3.402823466e38f;
    float mxx = -3.402823466e38f, mxy = -3.402823466e38f;
#pragma unroll
    for (int k = 0; k < 9; ++k) {
      float ox = ldsP[(1 + 2*k)*TP + tid] + b_off[2*k];
      float oy = ldsP[(2 + 2*k)*TP + tid] + b_off[2*k + 1];
      ox = expf(ox * scale) - 1.f;
      oy = expf(oy * scale) - 1.f;
      float cx = fminf(fmaxf(ox + xs, 0.f), img_w - 1.f);
      float cy = fminf(fmaxf(oy + ys, 0.f), img_h - 1.f);
      mnx = fminf(mnx, cx); mxx = fmaxf(mxx, cx);
      mny = fminf(mny, cy); mxy = fmaxf(mxy, cy);
    }
    float* o = out + ((size_t)n*HW + (size_t)h*Wf + pw)*5;
    o[0] = score; o[1] = mnx; o[2] = mny; o[3] = mxx; o[4] = mxy;
  }
}

// ---------- NMS: radix-select top-1000, bitonic sort, greedy suppress, compact ----------
__global__ __launch_bounds__(1024) void k_nms(const float* __restrict__ out,
                                              float* __restrict__ dout)
{
  __shared__ unsigned int hist[256];
  __shared__ unsigned int s_bin, s_need, s_total;
  __shared__ unsigned int counters[2];
  __shared__ float cs[CAND];
  __shared__ int   cidx[CAND];
  __shared__ float bx1[CAND], by1[CAND], bx2[CAND], by2[CAND], barea[CAND];
  __shared__ unsigned char sup[CAND];
  __shared__ unsigned int scanbuf[CAND];

  const int tid = threadIdx.x;
  const int n = blockIdx.x;
  const float* sc = out + (size_t)n*HW*5;

  // exact radix-select: key of the 1000th-largest score
  unsigned int prefix = 0, need = PRE_K;
  for (int round = 0; round < 4; ++round) {
    const int shift = 24 - 8*round;
    if (tid < 256) hist[tid] = 0u;
    __syncthreads();
    for (int i = tid; i < HW; i += 1024) {
      unsigned int k = fkey(sc[(size_t)i*5]);
      if (round == 0 || (k >> (shift + 8)) == prefix)
        atomicAdd(&hist[(k >> shift) & 255u], 1u);
    }
    __syncthreads();
    if (tid == 0) {
      unsigned int acc = 0; int b = 255;
      for (; b > 0; --b) { if (acc + hist[b] >= need) break; acc += hist[b]; }
      s_bin = (unsigned int)b; s_need = need - acc;
    }
    __syncthreads();
    prefix = (prefix << 8) | s_bin;
    need = s_need;
    __syncthreads();
  }
  const unsigned int T = prefix;
  const unsigned int nties = need;
  const unsigned int ngreat = PRE_K - nties;

  cs[tid] = -__builtin_inff();
  cidx[tid] = 0x7fffffff;
  if (tid < 2) counters[tid] = 0u;
  __syncthreads();
  for (int i = tid; i < HW; i += 1024) {
    float s = sc[(size_t)i*5];
    unsigned int k = fkey(s);
    int slot = -1;
    if (k > T) slot = (int)atomicAdd(&counters[0], 1u);
    else if (k == T) {
      unsigned int t = atomicAdd(&counters[1], 1u);
      if (t < nties) slot = (int)(ngreat + t);
    }
    if (slot >= 0) { cs[slot] = s; cidx[slot] = i; }
  }
  __syncthreads();

  // bitonic sort 1024 by (score desc, idx asc)
  for (int k = 2; k <= CAND; k <<= 1) {
    for (int j = k >> 1; j > 0; j >>= 1) {
      int p = tid ^ j;
      if (p > tid) {
        bool up = ((tid & k) == 0);
        float s1 = cs[tid], s2 = cs[p];
        int i1 = cidx[tid], i2 = cidx[p];
        bool ordered = (s1 > s2) || ((s1 == s2) && (i1 <= i2));
        if (ordered != up) { cs[tid] = s2; cs[p] = s1; cidx[tid] = i2; cidx[p] = i1; }
      }
      __syncthreads();
    }
  }

  { // gather boxes of sorted candidates
    int id = cidx[tid];
    float x1 = 0.f, y1 = 0.f, x2 = 0.f, y2 = 0.f;
    if (id < HW) {
      const float* bpx = out + ((size_t)n*HW + id)*5;
      x1 = bpx[1]; y1 = bpx[2]; x2 = bpx[3]; y2 = bpx[4];
    }
    bx1[tid] = x1; by1[tid] = y1; bx2[tid] = x2; by2[tid] = y2;
    barea[tid] = (x2 - x1) * (y2 - y1);
    sup[tid] = 0;
    if (tid >= PRE_K) cs[tid] = -__builtin_inff();
  }
  __syncthreads();

  // greedy suppression, score order
  for (int i = 0; i < PRE_K; ++i) {
    if (sup[i] == 0) {
      if (tid > i && tid < PRE_K) {
        float iw = fminf(bx2[i], bx2[tid]) - fmaxf(bx1[i], bx1[tid]);
        float ih = fminf(by2[i], by2[tid]) - fmaxf(by1[i], by1[tid]);
        iw = fmaxf(iw, 0.f); ih = fmaxf(ih, 0.f);
        float inter = iw * ih;
        float iou = inter / (barea[i] + barea[tid] - inter + 1e-9f);
        if (iou > 0.7f) sup[tid] = 1;
      }
    }
    __syncthreads();
  }

  // ordered compaction: alive first (sorted), then suppressed with -inf score
  unsigned int aflag = (tid < PRE_K && sup[tid] == 0) ? 1u : 0u;
  scanbuf[tid] = aflag;
  __syncthreads();
  for (int off = 1; off < CAND; off <<= 1) {
    unsigned int v = scanbuf[tid];
    unsigned int a = (tid >= off) ? scanbuf[tid - off] : 0u;
    __syncthreads();
    scanbuf[tid] = v + a;
    __syncthreads();
  }
  if (tid == CAND - 1) s_total = scanbuf[CAND - 1];
  unsigned int arank = scanbuf[tid] - aflag;
  __syncthreads();

  float* kb = dout + KB_OFF + (size_t)n*POST_K*4;
  float* ks = dout + KS_OFF + (size_t)n*POST_K;
  if (aflag && arank < POST_K) {
    kb[arank*4+0] = bx1[tid]; kb[arank*4+1] = by1[tid];
    kb[arank*4+2] = bx2[tid]; kb[arank*4+3] = by2[tid];
    ks[arank] = cs[tid];
  }

  unsigned int sflag = (tid < PRE_K && sup[tid] != 0) ? 1u : 0u;
  scanbuf[tid] = sflag;
  __syncthreads();
  for (int off = 1; off < CAND; off <<= 1) {
    unsigned int v = scanbuf[tid];
    unsigned int a = (tid >= off) ? scanbuf[tid - off] : 0u;
    __syncthreads();
    scanbuf[tid] = v + a;
    __syncthreads();
  }
  unsigned int srank = scanbuf[tid] - sflag;
  unsigned int slot = s_total + srank;
  if (sflag && slot < POST_K) {
    kb[slot*4+0] = bx1[tid]; kb[slot*4+1] = by1[tid];
    kb[slot*4+2] = bx2[tid]; kb[slot*4+3] = by2[tid];
    ks[slot] = -__builtin_inff();
  }
}

// ---------- launcher ----------
extern "C" void kernel_launch(void* const* d_in, const int* in_sizes, int n_in,
                              void* d_out, int out_size, void* d_ws, size_t ws_size,
                              hipStream_t stream)
{
  const float* x      = (const float*)d_in[0];
  const float* W_conv = (const float*)d_in[1];
  const float* b_conv = (const float*)d_in[2];
  const float* W_off  = (const float*)d_in[3];
  const float* b_off  = (const float*)d_in[4];
  const float* W_log  = (const float*)d_in[5];
  const float* b_log  = (const float*)d_in[6];
  const int*   image_h = (const int*)d_in[7];
  const int*   image_w = (const int*)d_in[8];

  const size_t xb_bytes = (size_t)Nimg*Hp*Wp*Cch*2;   // padded NHWC bf16
  const size_t wt_bytes = (size_t)Cch*Ktot*2;
  unsigned short* xbp = (unsigned short*)d_ws;
  unsigned short* Wt  = (unsigned short*)((char*)d_ws + xb_bytes);
  unsigned short* W19 = (unsigned short*)((char*)d_ws + xb_bytes + wt_bytes);

  float* out = (float*)d_out;

  const int tot_x = Nimg*Hp*Wp*Cch;
  k_prep_x  <<<(tot_x + 255)/256, 256, 0, stream>>>(x, xbp);
  k_prep_w  <<<(Cch*Ktot + 255)/256, 256, 0, stream>>>(W_conv, Wt);
  k_prep_w19<<<(32*Cch + 255)/256, 256, 0, stream>>>(W_log, W_off, W19);
  k_conv_fused<<<dim3(Wf/TP, Hf, Nimg), 256, 0, stream>>>(
      xbp, Wt, W19, b_conv, b_off, b_log, image_h, image_w, out);
  k_nms<<<Nimg, 1024, 0, stream>>>(out, out);
}
